// MoEReduceRSTensorParallel_61168924229674
// MI455X (gfx1250) — compile-verified, matrix operand order checked by reading, not run on hardware
//
#include <hip/hip_runtime.h>
#include <hip/hip_bf16.h>
#include <math.h>

// Problem constants (from reference)
#define NUM_TOKENS   2048
#define TOPK         2
#define NUM_ROWS     (NUM_TOKENS * TOPK)   // 4096 routed rows
#define NUM_EXPERTS  8
#define INTER        2048                  // K dimension of GEMM
#define HIDDEN       1024                  // N dimension of GEMM

// Tiling
#define MTILE    64         // per-block row tile (4 sub-tiles of 16)
#define TILE_N   128        // per-block column tile (8 waves x 16)
#define KSTEP    32         // bf16 WMMA K per step
#define APITCH   48         // LDS row pitch (bf16 elems): 96B rows, 16B-aligned slices
#define BPITCH   48

typedef __attribute__((ext_vector_type(16))) __bf16 v16bf;
typedef __attribute__((ext_vector_type(8)))  float  v8f;

union FragAB { uint4 u[2]; v16bf v; };
union FragC  { v8f v; float f[8]; };
union Pack8  { __bf16 h[8]; uint4 u; };
union Pack4  { __bf16 h[4]; uint2 u; };

// ---------------------------------------------------------------------------
// Kernel 1: routing — softmax over 8 experts, top-2 ids + weights per token.
// ---------------------------------------------------------------------------
__global__ __launch_bounds__(256) void routing_kernel(
    const float* __restrict__ logits,  // [T, 8]
    int*   __restrict__ expertOf,      // [T*2]
    float* __restrict__ wflat)         // [T*2]
{
    int t = blockIdx.x * blockDim.x + threadIdx.x;
    if (t >= NUM_TOKENS) return;
    float l[NUM_EXPERTS];
    float mx = -INFINITY;
    #pragma unroll
    for (int i = 0; i < NUM_EXPERTS; ++i) { l[i] = logits[t * NUM_EXPERTS + i]; mx = fmaxf(mx, l[i]); }
    float p[NUM_EXPERTS]; float s = 0.f;
    #pragma unroll
    for (int i = 0; i < NUM_EXPERTS; ++i) { p[i] = __expf(l[i] - mx); s += p[i]; }
    float inv = 1.0f / s;
    int i0 = 0; float p0 = p[0];
    #pragma unroll
    for (int i = 1; i < NUM_EXPERTS; ++i) if (p[i] > p0) { p0 = p[i]; i0 = i; }
    int i1 = -1; float p1 = -INFINITY;
    #pragma unroll
    for (int i = 0; i < NUM_EXPERTS; ++i) if (i != i0 && p[i] > p1) { p1 = p[i]; i1 = i; }
    expertOf[2 * t + 0] = i0;  wflat[2 * t + 0] = p0 * inv;
    expertOf[2 * t + 1] = i1;  wflat[2 * t + 1] = p1 * inv;
}

// ---------------------------------------------------------------------------
// Kernel 2: stable, deterministic per-expert bucketing (wave32 ballot scan).
// ---------------------------------------------------------------------------
__global__ __launch_bounds__(256) void bucketize_kernel(
    const int* __restrict__ expertOf,  // [R]
    int* __restrict__ bucket,          // [E, R]
    int* __restrict__ cnt)             // [E]
{
    int lane = threadIdx.x & 31;
    int e    = threadIdx.x >> 5;       // wave id == expert id
    unsigned c = 0;
    for (int base = 0; base < NUM_ROWS; base += 32) {
        int r = base + lane;
        bool pred = (expertOf[r] == e);
        unsigned mask = (unsigned)__ballot(pred);          // wave32: low 32 bits
        if (pred) {
            unsigned pos = c + __popc(mask & ((1u << lane) - 1u));
            bucket[e * NUM_ROWS + pos] = r;
        }
        c += __popc(mask);
    }
    if (lane == 0) cnt[e] = (int)c;
}

// ---------------------------------------------------------------------------
// Kernel 3: zero output (two atomic contributions per element accumulate in).
// ---------------------------------------------------------------------------
__global__ __launch_bounds__(256) void zero_kernel(float4* __restrict__ out4)
{
    out4[blockIdx.x * 256 + threadIdx.x] = make_float4(0.f, 0.f, 0.f, 0.f);
}

// ---------------------------------------------------------------------------
// Kernel 4: grouped expert GEMM, bf16 WMMA, f32 accumulate.
// Block = 256 thr (8 waves). Tile = 64 rows x 128 cols. Double-buffered LDS,
// software-pipelined K loop (1 barrier per K-step of 32).
// ---------------------------------------------------------------------------
__global__ __launch_bounds__(256) void moe_gemm_kernel(
    const float* __restrict__ X,       // [R, INTER]
    const float* __restrict__ W,       // [E, INTER, HIDDEN]
    const int*   __restrict__ bucket,  // [E, R]
    const int*   __restrict__ cnt,     // [E]
    const float* __restrict__ wflat,   // [R]
    float*       __restrict__ out)     // [T, HIDDEN]
{
    const int e       = blockIdx.z;
    const int rowTile = blockIdx.y;
    const int nRows   = cnt[e];
    if (rowTile * MTILE >= nRows) return;
    const int hBase = blockIdx.x * TILE_N;

    __shared__ __align__(16) __bf16 ldsA[2][MTILE * APITCH];
    __shared__ __align__(16) __bf16 ldsB[2][TILE_N * BPITCH];
    __shared__ int   sRow[MTILE];
    __shared__ float sW[MTILE];

    const int tid  = threadIdx.x;
    const int lane = tid & 31;
    const int wave = tid >> 5;

    if (tid < MTILE) {
        int idx = rowTile * MTILE + tid;
        int row = (idx < nRows) ? bucket[e * NUM_ROWS + idx] : -1;
        sRow[tid] = row;
        sW[tid]   = (row >= 0) ? wflat[row] : 0.0f;
    }
    __syncthreads();

    // A staging: thread -> (row mA, 8 consecutive k at kgA); one b128 LDS store.
    const int mA  = tid >> 2;              // 0..63
    const int kgA = (tid & 3) * 8;         // 0,8,16,24
    const int rowA = sRow[mA];
    const float* aSrc = (rowA >= 0) ? (X + (size_t)rowA * INTER + kgA) : nullptr;
    __bf16* const ldsAdst = &ldsA[0][0] + mA * APITCH + kgA;

    // B staging: thread -> (col nB, 16 consecutive k at khB); coalesced b32
    // column reads, packed to 4x ds_store_b64 into [n][k] layout.
    const int nB  = tid & (TILE_N - 1);    // 0..127
    const int khB = (tid >> 7) * 16;       // 0 or 16
    const float* bSrc = W + ((size_t)e * INTER) * HIDDEN + hBase + nB;
    __bf16* const ldsBdst = &ldsB[0][0] + nB * BPITCH + khB;

    // Fragment addresses (ISA 16-bit layouts): lane ml, half h:
    // elems 0-7 -> K=h*8+0..7 ; elems 8-15 -> K=16+h*8+0..7 (16B-aligned b128s)
    const int ml   = lane & 15;
    const int half = lane >> 4;
    const int aFragOff = ml * APITCH + half * 8;
    const int bFragOff = (wave * 16 + ml) * BPITCH + half * 8;

    FragC acc[4];
    #pragma unroll
    for (int i = 0; i < 4; ++i) acc[i].v = (v8f){0.f,0.f,0.f,0.f,0.f,0.f,0.f,0.f};

    float aR[8];
    float bR[16];

    auto loadGlobals = [&](int kk) {
        if (aSrc) {
            float4 x0 = *(const float4*)(aSrc + kk);
            float4 x1 = *(const float4*)(aSrc + kk + 4);
            aR[0]=x0.x; aR[1]=x0.y; aR[2]=x0.z; aR[3]=x0.w;
            aR[4]=x1.x; aR[5]=x1.y; aR[6]=x1.z; aR[7]=x1.w;
        } else {
            #pragma unroll
            for (int j = 0; j < 8; ++j) aR[j] = 0.f;
        }
        #pragma unroll
        for (int j = 0; j < 16; ++j)
            bR[j] = bSrc[(size_t)(kk + khB + j) * HIDDEN];
    };

    auto cvtStore = [&](int buf) {
        Pack8 pa;
        #pragma unroll
        for (int j = 0; j < 8; ++j) pa.h[j] = (__bf16)aR[j];
        *(uint4*)(ldsAdst + buf * (MTILE * APITCH)) = pa.u;
        #pragma unroll
        for (int g = 0; g < 4; ++g) {
            Pack4 pb;
            #pragma unroll
            for (int j = 0; j < 4; ++j) pb.h[j] = (__bf16)bR[4*g + j];
            *(uint2*)(ldsBdst + buf * (TILE_N * BPITCH) + 4*g) = pb.u;
        }
    };

    auto compute = [&](int buf) {
        const __bf16* sA = &ldsA[buf][0];
        const __bf16* sB = &ldsB[buf][0];
        FragAB b;
        b.u[0] = *(const uint4*)(sB + bFragOff);
        b.u[1] = *(const uint4*)(sB + bFragOff + 16);
        #pragma unroll
        for (int i = 0; i < 4; ++i) {
            FragAB a;
            a.u[0] = *(const uint4*)(sA + aFragOff + i * 16 * APITCH);
            a.u[1] = *(const uint4*)(sA + aFragOff + i * 16 * APITCH + 16);
            acc[i].v = __builtin_amdgcn_wmma_f32_16x16x32_bf16(
                false, a.v, false, b.v, (short)0, acc[i].v, false, false);
        }
    };

    // Prologue: stage kk=0 into buffer 0.
    loadGlobals(0);
    cvtStore(0);

    // Pipelined main loop: 2 K-steps per iteration (static double buffering).
    for (int kk = 0; kk < INTER; kk += 2 * KSTEP) {
        loadGlobals(kk + KSTEP);           // regs for buf1 (always in range)
        __syncthreads();                   // buf0 stores visible, buf1 reads done
        compute(0);
        cvtStore(1);
        bool hasNext = (kk + 2 * KSTEP) < INTER;
        if (hasNext) loadGlobals(kk + 2 * KSTEP);
        __syncthreads();                   // buf1 stores visible, buf0 reads done
        compute(1);
        if (hasNext) cvtStore(0);
    }

    // Epilogue: fold top-k weight; 2 commutative atomics per out element.
    const int col = hBase + wave * 16 + ml;
    float* outCol = out + col;
    #pragma unroll
    for (int i = 0; i < 4; ++i) {
        #pragma unroll
        for (int v = 0; v < 8; ++v) {
            int m   = i * 16 + v + half * 8;   // C/D layout: VGPR v, lanes16-31 -> M=8+v
            int row = sRow[m];
            if (row >= 0) {
                atomicAdd(outCol + (size_t)(row >> 1) * HIDDEN, acc[i].f[v] * sW[m]);
            }
        }
    }
}

// ---------------------------------------------------------------------------
extern "C" void kernel_launch(void* const* d_in, const int* in_sizes, int n_in,
                              void* d_out, int out_size, void* d_ws, size_t ws_size,
                              hipStream_t stream) {
    const float* X      = (const float*)d_in[0];  // [4096, 2048]
    const float* W      = (const float*)d_in[1];  // [8, 2048, 1024]
    const float* logits = (const float*)d_in[2];  // [2048, 8]
    float* out = (float*)d_out;                   // [2048, 1024]

    char* ws = (char*)d_ws;
    int*   expertOf = (int*)  (ws + 0);                         // 4096 ints
    float* wflat    = (float*)(ws + 16384);                     // 4096 floats
    int*   bucket   = (int*)  (ws + 32768);                     // 8*4096 ints
    int*   cnt      = (int*)  (ws + 32768 + 131072);            // 8 ints

    routing_kernel<<<NUM_TOKENS / 256, 256, 0, stream>>>(logits, expertOf, wflat);
    bucketize_kernel<<<1, 256, 0, stream>>>(expertOf, bucket, cnt);
    zero_kernel<<<(NUM_TOKENS * HIDDEN) / 1024, 256, 0, stream>>>((float4*)out);

    dim3 grid(HIDDEN / TILE_N, NUM_ROWS / MTILE, NUM_EXPERTS);  // (8, 64, 8)
    moe_gemm_kernel<<<grid, 256, 0, stream>>>(X, W, bucket, cnt, wflat, out);
}
